// SelfAttention_31834297598888
// MI455X (gfx1250) — matrix operand order repaired and checked by reading, not compiled, not run on hardware
//
#include <hip/hip_runtime.h>
#include <math.h>

// ---------------------------------------------------------------------------
// Self-attention forward for MI455X (gfx1250), wave32, WMMA bf16 path with
// async global->LDS staging (double buffered) in the attention kernel.
//   q = x Wq, k = x Wk, v = x Wv            [B,T,64]
//   S = q k^T * sqrt(64); P = softmax(S); out = P v
// ---------------------------------------------------------------------------

typedef __bf16 bf16;
typedef __attribute__((ext_vector_type(16))) bf16  v16bf;
typedef __attribute__((ext_vector_type(8)))  bf16  v8bf;
typedef __attribute__((ext_vector_type(8)))  float v8f;

static constexpr int Bn = 4;
static constexpr int Tn = 4096;
static constexpr int Cn = 512;
static constexpr int Hn = 64;
static constexpr int BT = Bn * Tn;          // 16384 rows total
static constexpr float SCALE = 8.0f;        // sqrt(64), multiplied as in ref

__device__ __forceinline__ v8f wmma_bf16(v16bf a, v16bf b, v8f c) {
  // D = A(16x32 bf16) * B(32x16 bf16) + C(16x16 f32)
  return __builtin_amdgcn_wmma_f32_16x16x32_bf16(
      /*neg_a=*/false, a, /*neg_b=*/false, b,
      /*c_mod=*/(short)0, c, /*reuse_a=*/false, /*reuse_b=*/false);
}

__device__ __forceinline__ v8f zero8() {
  v8f z;
#pragma unroll
  for (int i = 0; i < 8; ++i) z[i] = 0.0f;
  return z;
}

// Async 16B global -> LDS copy (per lane), tracked by ASYNCcnt.
//   lds_off : LDS byte offset (low 32 bits of the generic shared pointer)
//   voff    : per-lane unsigned byte offset from sbase
//   sbase   : uniform 64-bit global byte address (SGPR pair)
__device__ __forceinline__ void async_load_16B(unsigned int lds_off,
                                               unsigned int voff,
                                               unsigned long long sbase) {
  asm volatile("global_load_async_to_lds_b128 %0, %1, %2"
               :: "v"(lds_off), "v"(voff), "s"(sbase) : "memory");
}

__device__ __forceinline__ unsigned int lds_addr(const void* p) {
  return (unsigned int)(unsigned long long)(uintptr_t)p;  // addr[31:0] = LDS offset
}

// ---------------------------------------------------------------------------
// Kernel 0: transpose the three [512,64] weight matrices into WT[3][64][512]
// so that kernel 1 can read B-operand columns contiguously.
// ---------------------------------------------------------------------------
__global__ void wt_transpose_kernel(const float* __restrict__ Wq,
                                    const float* __restrict__ Wk,
                                    const float* __restrict__ Wv,
                                    float* __restrict__ WT) {
  int idx = blockIdx.x * blockDim.x + threadIdx.x;
  if (idx >= 3 * Cn * Hn) return;
  int mat = idx / (Cn * Hn);
  int rem = idx - mat * (Cn * Hn);
  int r = rem / Hn;        // 0..511  (C index)
  int c = rem - r * Hn;    // 0..63   (H index)
  const float* W = (mat == 0) ? Wq : ((mat == 1) ? Wk : Wv);
  WT[((size_t)mat * Hn + c) * Cn + r] = W[(size_t)r * Hn + c];
}

// ---------------------------------------------------------------------------
// Kernel 1: QKV projection via WMMA.
//   grid.x = BT/64 (64 rows per WG, 4 waves x 16 rows), grid.y = matrix id.
//   Outputs: Q,K bf16 row-major [BT,64]; V bf16 transposed [B,64,T].
// ---------------------------------------------------------------------------
__global__ __launch_bounds__(128) void qkv_kernel(
    const float* __restrict__ x,   // [BT, 512]
    const float* __restrict__ WT,  // [3, 64, 512]
    bf16* __restrict__ QB, bf16* __restrict__ KB, bf16* __restrict__ VT) {
  const int tid  = threadIdx.x;
  const int wave = tid >> 5;
  const int lane = tid & 31;
  const int hl   = lane >> 4;    // half-wave (0/1)
  const int lr   = lane & 15;

  const int g0  = blockIdx.x * 64;     // row base in [0, BT)
  const int mat = blockIdx.y;          // 0=q 1=k 2=v
  const float* Wt   = WT + (size_t)mat * Hn * Cn;
  const float* xrow = x + (size_t)(g0 + wave * 16 + lr) * Cn;

  v8f acc[4];
#pragma unroll
  for (int nt = 0; nt < 4; ++nt) acc[nt] = zero8();

  for (int kk = 0; kk < Cn; kk += 32) {
    // ---- A operand: 16x32 tile of x rows, fp32 -> bf16.
    // lanes 0-15 hold K 0..15; lanes 16-31 hold K 16..31.
    v16bf Ax;
    {
      const float* p = xrow + kk + (hl ? 16 : 0);
#pragma unroll
      for (int i = 0; i < 8; ++i) {
        Ax[i]     = (bf16)p[i];
        Ax[8 + i] = (bf16)p[8 + i];
      }
    }
    // ---- B operands: 32x16 tiles of W (column n contiguous via WT).
#pragma unroll
    for (int nt = 0; nt < 4; ++nt) {
      const float* wp = Wt + (size_t)(nt * 16 + lr) * Cn + kk + hl * 16;
      v16bf Bw;
#pragma unroll
      for (int i = 0; i < 16; ++i) Bw[i] = (bf16)wp[i];
      acc[nt] = wmma_bf16(Ax, Bw, acc[nt]);
    }
  }

  // ---- store: C layout is row M = j + 8*hl (VGPR j), col N = lane&15.
#pragma unroll
  for (int nt = 0; nt < 4; ++nt) {
#pragma unroll
    for (int j = 0; j < 8; ++j) {
      const int row = g0 + wave * 16 + j + 8 * hl;   // global row in [0,BT)
      const int col = nt * 16 + lr;
      const bf16 val = (bf16)acc[nt][j];
      if (mat == 0) {
        QB[(size_t)row * Hn + col] = val;
      } else if (mat == 1) {
        KB[(size_t)row * Hn + col] = val;
      } else {
        const int bb = row / Tn;
        const int tt = row - bb * Tn;
        VT[((size_t)bb * Hn + col) * Tn + tt] = val;   // transposed V
      }
    }
  }
}

// ---------------------------------------------------------------------------
// Kernel 2: flash-attention over all T keys (no mask), 64 queries per WG.
// K/V blocks staged in LDS with global_load_async_to_lds_b128, double
// buffered; B-operands for both GEMMs read from LDS (ds_load_b128).
// ---------------------------------------------------------------------------
__global__ __launch_bounds__(128) void attn_kernel(
    const bf16* __restrict__ QB,   // [BT, 64]
    const bf16* __restrict__ KB,   // [BT, 64]
    const bf16* __restrict__ VT,   // [B, 64, T]
    float* __restrict__ out) {     // [BT, 64]
  const int tid  = threadIdx.x;
  const int wave = tid >> 5;
  const int lane = tid & 31;
  const int hl   = lane >> 4;
  const int lr   = lane & 15;

  const int g0 = blockIdx.x * 64;            // query row base in [0, BT)
  const int b  = g0 / Tn;
  const bf16* Kb = KB + (size_t)b * Tn * Hn;
  const bf16* Vb = VT + (size_t)b * Hn * Tn;

  // double-buffered K/V tiles + per-wave private P staging
  __shared__ __align__(64) bf16 Klds[2][64 * 64];   // [key][hd]
  __shared__ __align__(64) bf16 Vlds[2][64 * 64];   // [hd][key]
  __shared__ __align__(64) bf16 Plds[4][16 * 64];
  bf16* Pw = Plds[wave];

  const unsigned long long kbase = (unsigned long long)(uintptr_t)Kb;
  const unsigned long long vbase = (unsigned long long)(uintptr_t)Vb;
  const unsigned int klds0 = lds_addr(&Klds[0][0]);
  const unsigned int klds1 = lds_addr(&Klds[1][0]);
  const unsigned int vlds0 = lds_addr(&Vlds[0][0]);
  const unsigned int vlds1 = lds_addr(&Vlds[1][0]);

  // Issue one 64-key block prefetch (8KB K + 8KB V per WG; 8 b128/lane-slot).
  auto prefetch = [&](int kb, int buf) {
    const unsigned int kl = buf ? klds1 : klds0;
    const unsigned int vl = buf ? vlds1 : vlds0;
#pragma unroll
    for (int i = 0; i < 4; ++i) {
      const unsigned int c = (unsigned int)(tid + 128 * i);  // chunk 0..511
      const unsigned int l = c * 16u;                        // LDS byte offset
      // K block: 64 rows x 128B, contiguous at byte offset kb*128
      async_load_16B(kl + l, (unsigned int)kb * 128u + l, kbase);
      // V block: 64 hd-rows x 128B, row stride Tn*2 bytes
      const unsigned int row = c >> 3;
      const unsigned int col = (c & 7u) * 16u;
      async_load_16B(vl + l, row * (unsigned int)(Tn * 2) +
                             (unsigned int)kb * 2u + col, vbase);
    }
  };

  // ---- load Q tile once, already in WMMA A layout (2 K-steps over H=64)
  const bf16* qrow = QB + (size_t)(g0 + wave * 16 + lr) * Hn;
  v16bf Aq[2];
#pragma unroll
  for (int ks = 0; ks < 2; ++ks) {
    const bf16* p = qrow + ks * 32 + (hl ? 16 : 0);
    v8bf lo = *(const v8bf*)p;
    v8bf hi = *(const v8bf*)(p + 8);
#pragma unroll
    for (int i = 0; i < 8; ++i) { Aq[ks][i] = lo[i]; Aq[ks][8 + i] = hi[i]; }
  }

  v8f Oacc[4];
#pragma unroll
  for (int nt = 0; nt < 4; ++nt) Oacc[nt] = zero8();
  float m[8], l[8];
#pragma unroll
  for (int j = 0; j < 8; ++j) { m[j] = -INFINITY; l[j] = 0.0f; }

  prefetch(0, 0);                                    // block 0 -> buf 0

  constexpr int NB = Tn / 64;                        // 64 key blocks
  for (int i = 0; i < NB; ++i) {
    const int kb = i * 64;
    const bool has_next = (i + 1) < NB;
    if (has_next) prefetch(kb + 64, (i + 1) & 1);    // 8 more in flight

    // current block's 8 per-wave async loads are the oldest; async loads
    // complete in order, so asynccnt<=8 (or 0 on last block) retires them.
    if (has_next) asm volatile("s_wait_asynccnt 0x8" ::: "memory");
    else          asm volatile("s_wait_asynccnt 0x0" ::: "memory");
    __syncthreads();                                 // all waves' data landed

    const bf16* Kc = Klds[i & 1];
    const bf16* Vc = Vlds[i & 1];

    // ---------------- S = (Q K^T) * SCALE, 16q x 64k per wave --------------
    v8f S[4];
#pragma unroll
    for (int nt = 0; nt < 4; ++nt) {
      S[nt] = zero8();
#pragma unroll
      for (int ks = 0; ks < 2; ++ks) {
        v16bf Bk = *(const v16bf*)(Kc + (nt * 16 + lr) * 64 + ks * 32 + hl * 16);
        S[nt] = wmma_bf16(Aq[ks], Bk, S[nt]);
      }
#pragma unroll
      for (int j = 0; j < 8; ++j) S[nt][j] *= SCALE;
    }

    // ---------------- online softmax (rows live in 16-lane halves) --------
    float alpha[8];
#pragma unroll
    for (int j = 0; j < 8; ++j) {
      float rm = fmaxf(fmaxf(S[0][j], S[1][j]), fmaxf(S[2][j], S[3][j]));
#pragma unroll
      for (int off = 8; off >= 1; off >>= 1)
        rm = fmaxf(rm, __shfl_xor(rm, off, 32));
      const float mn = fmaxf(m[j], rm);
      alpha[j] = __expf(m[j] - mn);
      m[j] = mn;
    }

    float rsum[8];
#pragma unroll
    for (int j = 0; j < 8; ++j) rsum[j] = 0.0f;
#pragma unroll
    for (int nt = 0; nt < 4; ++nt) {
#pragma unroll
      for (int j = 0; j < 8; ++j) {
        const float p = __expf(S[nt][j] - m[j]);
        rsum[j] += p;
        Pw[(j + 8 * hl) * 64 + nt * 16 + lr] = (bf16)p;
      }
    }
#pragma unroll
    for (int j = 0; j < 8; ++j) {
      float rs = rsum[j];
#pragma unroll
      for (int off = 8; off >= 1; off >>= 1) rs += __shfl_xor(rs, off, 32);
      l[j] = l[j] * alpha[j] + rs;
#pragma unroll
      for (int nt = 0; nt < 4; ++nt) Oacc[nt][j] *= alpha[j];
    }

    asm volatile("" ::: "memory");   // keep per-wave DS stores before DS loads

    // ---------------- O += P V  (A = P from LDS, B = V^T columns) ---------
#pragma unroll
    for (int ks = 0; ks < 2; ++ks) {
      const bf16* pp = Pw + lr * 64 + ks * 32 + (hl ? 16 : 0);
      v8bf lo = *(const v8bf*)pp;
      v8bf hi = *(const v8bf*)(pp + 8);
      v16bf Ap;
#pragma unroll
      for (int ii = 0; ii < 8; ++ii) { Ap[ii] = lo[ii]; Ap[8 + ii] = hi[ii]; }
#pragma unroll
      for (int nt = 0; nt < 4; ++nt) {
        v16bf Bv = *(const v16bf*)(Vc + (nt * 16 + lr) * 64 + ks * 32 + hl * 16);
        Oacc[nt] = wmma_bf16(Ap, Bv, Oacc[nt]);
      }
    }

    __syncthreads();   // everyone done with buf[i&1] before it is re-filled
  }

  // ---------------- epilogue: normalize and store fp32 ---------------------
#pragma unroll
  for (int j = 0; j < 8; ++j) {
    const float inv = 1.0f / l[j];
    const int row = g0 + wave * 16 + j + 8 * hl;
#pragma unroll
    for (int nt = 0; nt < 4; ++nt) {
      out[(size_t)row * Hn + nt * 16 + lr] = Oacc[nt][j] * inv;
    }
  }
}

// ---------------------------------------------------------------------------
// Host-side launcher (all launches on `stream`; scratch from d_ws only).
// ---------------------------------------------------------------------------
extern "C" void kernel_launch(void* const* d_in, const int* in_sizes, int n_in,
                              void* d_out, int out_size, void* d_ws, size_t ws_size,
                              hipStream_t stream) {
  const float* x  = (const float*)d_in[0];   // [4,4096,512]
  const float* Wq = (const float*)d_in[1];   // [512,64]
  const float* Wk = (const float*)d_in[2];
  const float* Wv = (const float*)d_in[3];
  float* out = (float*)d_out;                // [4,4096,64]

  // workspace layout (bytes):
  //   [0,            2MB)  QB  bf16 [BT,64]
  //   [2MB,          4MB)  KB  bf16 [BT,64]
  //   [4MB,          6MB)  VT  bf16 [B,64,T]
  //   [6MB,  6MB+384KB)    WT  f32  [3,64,512]
  char* ws = (char*)d_ws;
  const size_t qkv_bytes = (size_t)BT * Hn * sizeof(bf16);   // 2 MB each
  bf16*  QB = (bf16*)(ws);
  bf16*  KB = (bf16*)(ws + qkv_bytes);
  bf16*  VT = (bf16*)(ws + 2 * qkv_bytes);
  float* WT = (float*)(ws + 3 * qkv_bytes);

  (void)in_sizes; (void)n_in; (void)out_size; (void)ws_size;

  // 0) transpose weights
  {
    const int n = 3 * Cn * Hn;
    wt_transpose_kernel<<<(n + 255) / 256, 256, 0, stream>>>(Wq, Wk, Wv, WT);
  }
  // 1) QKV projection (y selects q/k/v)
  qkv_kernel<<<dim3(BT / 64, 3), 128, 0, stream>>>(x, WT, QB, KB, VT);
  // 2) flash attention
  attn_kernel<<<BT / 64, 128, 0, stream>>>(QB, KB, VT, out);
}
